// MiniGPT_64192581206028
// MI455X (gfx1250) — compile-verified, matrix operand order checked by reading, not run on hardware
//
#include <hip/hip_runtime.h>

// MiniGPT forward for MI455X (gfx1250, wave32, WMMA, async global->LDS).
// bf16 weights (~167MB, ~fits 192MB L2), fp32 residual stream, all GEMMs on
// V_WMMA_F32_16X16X32_BF16 with fp32 accumulation, flash attention on WMMA.
// Global->LDS staging uses GLOBAL_LOAD_ASYNC_TO_LDS_B128 (ASYNCcnt) with
// double-buffered LDS tiles so tile k+1 streams while tile k is in the
// matrix pipe.

#define WG_B 4
#define WG_T 1024
#define WG_D 1024
#define WG_H 16
#define WG_HD 64
#define WG_FF 4096
#define WG_L 4
#define WG_VOCAB 32000

typedef __bf16 bf16;
typedef __attribute__((ext_vector_type(16))) __bf16 v16bf;
typedef __attribute__((ext_vector_type(8)))  __bf16 v8bf;
typedef __attribute__((ext_vector_type(8)))  float  v8f;

__device__ __forceinline__ v8f wmma_bf(v16bf a, v16bf b, v8f c) {
  // (neg_a, A, neg_b, B, c_mod, C, reuse_a, reuse_b)
  return __builtin_amdgcn_wmma_f32_16x16x32_bf16(false, a, false, b, (short)0, c,
                                                 false, false);
}

__device__ __forceinline__ v8f zero8() {
  v8f z;
#pragma unroll
  for (int i = 0; i < 8; ++i) z[i] = 0.0f;
  return z;
}

// Low 32 bits of a generic LDS address are the offset within the group's LDS
// allocation (flat-aperture rule).
__device__ __forceinline__ unsigned lds_off(const void* p) {
  return (unsigned)(uintptr_t)p;
}

// Async DMA of 16 bytes global -> LDS (per lane). Tracked by ASYNCcnt; data
// never touches VGPRs. GV addressing: vdst = LDS byte offset, vaddr = 64-bit
// global address pair.
__device__ __forceinline__ void async_b128(unsigned dst_lds, const void* gsrc) {
  asm volatile("global_load_async_to_lds_b128 %0, %1, off"
               :: "v"(dst_lds), "v"(gsrc)
               : "memory");
}

__device__ __forceinline__ void wait_async0() {
  asm volatile("s_wait_asynccnt 0x0" ::: "memory");
}

// 16-bit A/B fragment (16 rows x 32 K) from a row-major LDS tile.
// ISA layout: lanes 0-15 hold M=lane with K={0..7,16..23}; lanes 16-31 hold
// M=lane-16 with K={8..15,24..31}; 2 bf16 per VGPR.
__device__ __forceinline__ v16bf load_frag(const bf16* base, int ld) {
  const int lane = threadIdx.x & 31;
  const int m  = lane & 15;
  const int kb = (lane >> 4) << 3;
  const bf16* r = base + m * ld;
  v16bf f;
#pragma unroll
  for (int v = 0; v < 8; ++v) {
    const int kk = (v < 4) ? (kb + 2 * v) : (16 + kb + 2 * (v - 4));
    f[2 * v]     = r[kk];
    f[2 * v + 1] = r[kk + 1];
  }
  return f;
}

// B fragment whose matrix columns are the tile's columns (K runs down rows):
// lane n holds tile column n, K packed down the rows.
__device__ __forceinline__ v16bf load_fragT(const bf16* base, int ld) {
  const int lane = threadIdx.x & 31;
  const int n  = lane & 15;
  const int kb = (lane >> 4) << 3;
  v16bf f;
#pragma unroll
  for (int v = 0; v < 8; ++v) {
    const int kk = (v < 4) ? (kb + 2 * v) : (16 + kb + 2 * (v - 4));
    f[2 * v]     = base[kk * ld + n];
    f[2 * v + 1] = base[(kk + 1) * ld + n];
  }
  return f;
}

// ---------------------------------------------------------------------------
// fp32 -> bf16 weight conversion (grid-stride)
__global__ void f2bf_kernel(const float* __restrict__ src, bf16* __restrict__ dst,
                            size_t n) {
  for (size_t i = (size_t)blockIdx.x * blockDim.x + threadIdx.x; i < n;
       i += (size_t)gridDim.x * blockDim.x)
    dst[i] = (bf16)src[i];
}

// ---------------------------------------------------------------------------
// Embedding: h[b*T+t, d] = tok_emb[x[b,t], d] + pos_emb[t, d]  (fp32)
__global__ void embed_kernel(const int* __restrict__ x, const float* __restrict__ te,
                             const float* __restrict__ pe, float* __restrict__ h) {
  const int row = blockIdx.x;                // b*T + t
  const int tok = x[row];
  const int tt  = row & (WG_T - 1);
  const float* tr = te + (size_t)tok * WG_D;
  const float* pr = pe + (size_t)tt * WG_D;
  float* hr = h + (size_t)row * WG_D;
  for (int d = threadIdx.x; d < WG_D; d += blockDim.x) hr[d] = tr[d] + pr[d];
}

// ---------------------------------------------------------------------------
// LayerNorm over D=1024, fp32 in -> bf16 out. One block (256 thr) per row.
__global__ void ln_kernel(const float* __restrict__ x, const float* __restrict__ g,
                          const float* __restrict__ bt, bf16* __restrict__ y) {
  __shared__ float ssum[256];
  __shared__ float ssq[256];
  const int row = blockIdx.x;
  const float* xr = x + (size_t)row * WG_D;
  float s = 0.f, q = 0.f;
  for (int d = threadIdx.x; d < WG_D; d += 256) {
    const float v = xr[d];
    s += v; q += v * v;
  }
  ssum[threadIdx.x] = s; ssq[threadIdx.x] = q;
  __syncthreads();
  for (int st = 128; st > 0; st >>= 1) {
    if (threadIdx.x < st) {
      ssum[threadIdx.x] += ssum[threadIdx.x + st];
      ssq[threadIdx.x]  += ssq[threadIdx.x + st];
    }
    __syncthreads();
  }
  const float mu  = ssum[0] * (1.f / WG_D);
  const float var = ssq[0] * (1.f / WG_D) - mu * mu;
  const float rs  = rsqrtf(var + 1e-5f);
  bf16* yr = y + (size_t)row * WG_D;
  for (int d = threadIdx.x; d < WG_D; d += 256)
    yr[d] = (bf16)((xr[d] - mu) * rs * g[d] + bt[d]);
}

// ---------------------------------------------------------------------------
// Tiled bf16 GEMM: C[M,N] = A[M,K] * W[N,K]^T (+bias, +GELU, +fp32 residual).
// Block 128x128, BK=32, 256 threads = 8 waves; wave tile 64x32 = 4x2 WMMA.
// Double-buffered LDS, async global->LDS copies pipelined against WMMA.
__global__ __launch_bounds__(256) void gemm_kernel(
    const bf16* __restrict__ A, const bf16* __restrict__ W,
    const float* __restrict__ bias, float* __restrict__ resid,
    bf16* __restrict__ outb, float* __restrict__ outf,
    int M, int N, int K, int gelu_flag) {
  __shared__ __align__(16) bf16 lA[2][128][40];   // +8 pad, 80B row stride
  __shared__ __align__(16) bf16 lB[2][128][40];
  const int t    = threadIdx.x;
  const int lane = t & 31;
  const int wave = t >> 5;
  const int wrow = wave >> 2;        // 0..1  (64 rows each)
  const int wcol = wave & 3;         // 0..3  (32 cols each)
  const int bm = blockIdx.y * 128;
  const int bn = blockIdx.x * 128;

  v8f acc[4][2];
#pragma unroll
  for (int i = 0; i < 4; ++i)
#pragma unroll
    for (int j = 0; j < 2; ++j) acc[i][j] = zero8();

  const int li = t >> 1;             // 0..127: tile row this thread copies
  const int lc = (t & 1) << 4;       // 0 or 16: first col of its 16-elem run

  const unsigned la[2] = { lds_off(&lA[0][li][lc]), lds_off(&lA[1][li][lc]) };
  const unsigned lb[2] = { lds_off(&lB[0][li][lc]), lds_off(&lB[1][li][lc]) };
  const bf16* gA = &A[(size_t)(bm + li) * K + lc];
  const bf16* gW = &W[(size_t)(bn + li) * K + lc];

  const int nk = K >> 5;
  // Prologue: tile 0 -> buffer 0 (4 async b128 copies per thread).
  async_b128(la[0],      gA);
  async_b128(la[0] + 16, gA + 8);
  async_b128(lb[0],      gW);
  async_b128(lb[0] + 16, gW + 8);

  for (int kt = 0; kt < nk; ++kt) {
    wait_async0();                   // tile kt landed in LDS (this wave's ops)
    __syncthreads();                 // ... and everyone's; all reads of nxt done
    const int cur = kt & 1;
    if (kt + 1 < nk) {               // stream tile kt+1 into the other buffer
      const int nxt = cur ^ 1;
      const bf16* na = gA + (size_t)(kt + 1) * 32;
      const bf16* nw = gW + (size_t)(kt + 1) * 32;
      async_b128(la[nxt],      na);
      async_b128(la[nxt] + 16, na + 8);
      async_b128(lb[nxt],      nw);
      async_b128(lb[nxt] + 16, nw + 8);
    }
    v16bf af[4], bfr[2];
#pragma unroll
    for (int i = 0; i < 4; ++i)
      af[i] = load_frag(&lA[cur][wrow * 64 + i * 16][0], 40);
#pragma unroll
    for (int j = 0; j < 2; ++j)
      bfr[j] = load_frag(&lB[cur][wcol * 32 + j * 16][0], 40);
#pragma unroll
    for (int i = 0; i < 4; ++i)
#pragma unroll
      for (int j = 0; j < 2; ++j) acc[i][j] = wmma_bf(af[i], bfr[j], acc[i][j]);
    __syncthreads();                 // done reading buf cur before it's refilled
  }

  // Epilogue. C frag layout: VGPR r -> row (lane>=16 ? 8 : 0)+r, col lane&15.
  const int mhalf = (lane >> 4) << 3;
  const int nlane = lane & 15;
#pragma unroll
  for (int i = 0; i < 4; ++i)
#pragma unroll
    for (int j = 0; j < 2; ++j)
#pragma unroll
      for (int r = 0; r < 8; ++r) {
        const int m = bm + wrow * 64 + i * 16 + mhalf + r;
        const int n = bn + wcol * 32 + j * 16 + nlane;
        float v = acc[i][j][r];
        if (bias) v += bias[n];
        if (gelu_flag) v = 0.5f * v * (1.f + erff(v * 0.70710678118654752f));
        const size_t o = (size_t)m * N + n;
        if (resid) { v += resid[o]; resid[o] = v; }
        if (outb) outb[o] = (bf16)v;
        if (outf) outf[o] = v;
      }
}

// ---------------------------------------------------------------------------
// Split GEMM output [4096, 3*D] into q/k/v laid out [B,H,T,HD] (bf16).
__global__ void qkv_split_kernel(const bf16* __restrict__ qkv, bf16* __restrict__ q,
                                 bf16* __restrict__ k, bf16* __restrict__ v) {
  const size_t i = (size_t)blockIdx.x * 256 + threadIdx.x;   // 4096*3072 total
  const int col = (int)(i % (3 * WG_D));
  const int row = (int)(i / (3 * WG_D));
  const int s  = col >> 10;          // 0:q 1:k 2:v
  const int h  = (col >> 6) & 15;
  const int hd = col & 63;
  const int b  = row >> 10;
  const int tt = row & 1023;
  const size_t o = (((size_t)(b * WG_H + h)) * WG_T + tt) * WG_HD + hd;
  const bf16 val = qkv[i];
  if (s == 0) q[o] = val; else if (s == 1) k[o] = val; else v[o] = val;
}

// ---------------------------------------------------------------------------
// Flash attention (causal), all-WMMA. grid = (T/128, B*H), block = 256 (8 waves).
// Each wave owns a 16-row Q strip; block streams 32-key K/V tiles via async LDS.
__global__ __launch_bounds__(256) void attn_kernel(
    const bf16* __restrict__ qg, const bf16* __restrict__ kg,
    const bf16* __restrict__ vg, bf16* __restrict__ out) {
  __shared__ __align__(16) bf16 lK[32][72];       // 32 keys x 64 d (+8 pad)
  __shared__ __align__(16) bf16 lV[32][72];
  __shared__ __align__(16) bf16 lP[8][16][40];    // per-wave P staging
  const int t = threadIdx.x, lane = t & 31, wave = t >> 5;
  const int bh = blockIdx.y;                      // b*H + h
  const int b = bh >> 4, h = bh & 15;
  const int qb = blockIdx.x * 128;
  const int q0 = qb + wave * 16;
  const bf16* qp = qg + (size_t)bh * (WG_T * WG_HD);
  const bf16* kp = kg + (size_t)bh * (WG_T * WG_HD);
  const bf16* vp = vg + (size_t)bh * (WG_T * WG_HD);

  const int nlane = lane & 15;
  const int kb8   = (lane >> 4) << 3;             // K-pack base AND row-half

  // Q A-fragments (16 rows x 64 d = two K=32 fragments), loaded once.
  v16bf qa[2];
#pragma unroll
  for (int hb = 0; hb < 2; ++hb) {
    const bf16* r = qp + (size_t)(q0 + nlane) * WG_HD + hb * 32;
#pragma unroll
    for (int v = 0; v < 8; ++v) {
      const int kk = (v < 4) ? (kb8 + 2 * v) : (16 + kb8 + 2 * (v - 4));
      qa[hb][2 * v]     = r[kk];
      qa[hb][2 * v + 1] = r[kk + 1];
    }
  }

  // Per-thread async-copy slots for the K/V tiles (8 bf16 = 16B each).
  const int ti = t >> 3, tc = (t & 7) << 3;
  const unsigned lko = lds_off(&lK[ti][tc]);
  const unsigned lvo = lds_off(&lV[ti][tc]);

  v8f o0 = zero8(), o1 = zero8(), o2 = zero8(), o3 = zero8();
  float mr[8], lr[8];
#pragma unroll
  for (int r = 0; r < 8; ++r) { mr[r] = -1e30f; lr[r] = 0.f; }

  const float scale = 0.125f;                     // 1/sqrt(HD=64)
  const int nkb = (qb + 128) >> 5;                // 32-key blocks this block needs
  for (int kt = 0; kt < nkb; ++kt) {
    const int kbase = kt << 5;
    __syncthreads();                              // prior reads of lK/lV done
    async_b128(lko, &kp[(size_t)(kbase + ti) * WG_HD + tc]);
    async_b128(lvo, &vp[(size_t)(kbase + ti) * WG_HD + tc]);
    wait_async0();
    __syncthreads();
    if (kbase <= q0 + 15) {                       // wave-uniform causal gate
      // S = Q.K^T for two 16-key groups (B columns are K rows -> load_frag)
      v8f s0 = zero8(), s1 = zero8();
      s0 = wmma_bf(qa[0], load_frag(&lK[0][0], 72),  s0);
      s0 = wmma_bf(qa[1], load_frag(&lK[0][32], 72), s0);
      s1 = wmma_bf(qa[0], load_frag(&lK[16][0], 72),  s1);
      s1 = wmma_bf(qa[1], load_frag(&lK[16][32], 72), s1);

      // online softmax; row stats live redundantly on each 16-lane half
#pragma unroll
      for (int r = 0; r < 8; ++r) {
        const int qi = q0 + kb8 + r;
        float a0 = s0[r] * scale;
        float a1 = s1[r] * scale;
        a0 = (kbase + nlane > qi)      ? -1e30f : a0;
        a1 = (kbase + 16 + nlane > qi) ? -1e30f : a1;
        float rmax = fmaxf(a0, a1);
#pragma unroll
        for (int d = 1; d < 16; d <<= 1) rmax = fmaxf(rmax, __shfl_xor(rmax, d, 16));
        const float mnew = fmaxf(mr[r], rmax);
        const float corr = __expf(mr[r] - mnew);
        const float p0 = __expf(a0 - mnew);
        const float p1 = __expf(a1 - mnew);
        float rsum = p0 + p1;
#pragma unroll
        for (int d = 1; d < 16; d <<= 1) rsum += __shfl_xor(rsum, d, 16);
        lr[r] = lr[r] * corr + rsum;
        mr[r] = mnew;
        o0[r] *= corr; o1[r] *= corr; o2[r] *= corr; o3[r] *= corr;
        lP[wave][kb8 + r][nlane]      = (bf16)p0;   // C-layout -> LDS
        lP[wave][kb8 + r][nlane + 16] = (bf16)p1;
      }
      // per-wave LDS ordering: ds ops are in-order per wave; make it airtight
      asm volatile("s_wait_dscnt 0x0" ::: "memory");
      const v16bf pa = load_frag(&lP[wave][0][0], 40);   // A-layout reload
      o0 = wmma_bf(pa, load_fragT(&lV[0][0],  72), o0);  // P.V, HD split in 4
      o1 = wmma_bf(pa, load_fragT(&lV[0][16], 72), o1);
      o2 = wmma_bf(pa, load_fragT(&lV[0][32], 72), o2);
      o3 = wmma_bf(pa, load_fragT(&lV[0][48], 72), o3);
    }
  }

  // Normalize and write bf16 [B*T, D] with col = h*64 + hd.
#pragma unroll
  for (int r = 0; r < 8; ++r) {
    const float inv = 1.0f / lr[r];
    const int qi = q0 + kb8 + r;
    const size_t base = ((size_t)(b * WG_T + qi)) * WG_D + h * WG_HD;
    out[base + 0  + nlane] = (bf16)(o0[r] * inv);
    out[base + 16 + nlane] = (bf16)(o1[r] * inv);
    out[base + 32 + nlane] = (bf16)(o2[r] * inv);
    out[base + 48 + nlane] = (bf16)(o3[r] * inv);
  }
}

// ---------------------------------------------------------------------------
extern "C" void kernel_launch(void* const* d_in, const int* in_sizes, int n_in,
                              void* d_out, int out_size, void* d_ws, size_t ws_size,
                              hipStream_t stream) {
  const int*   x       = (const int*)  d_in[0];
  const float* tok_emb = (const float*)d_in[1];
  const float* pos_emb = (const float*)d_in[2];
  const float* ln1_g   = (const float*)d_in[3];
  const float* ln1_b   = (const float*)d_in[4];
  const float* qkv_w   = (const float*)d_in[5];
  const float* qkv_b   = (const float*)d_in[6];
  const float* proj_w  = (const float*)d_in[7];
  const float* proj_b  = (const float*)d_in[8];
  const float* ln2_g   = (const float*)d_in[9];
  const float* ln2_b   = (const float*)d_in[10];
  const float* fc1_w   = (const float*)d_in[11];
  const float* fc1_b   = (const float*)d_in[12];
  const float* fc2_w   = (const float*)d_in[13];
  const float* fc2_b   = (const float*)d_in[14];
  const float* lnf_g   = (const float*)d_in[15];
  const float* lnf_b   = (const float*)d_in[16];
  float* logits = (float*)d_out;

  const int M = WG_B * WG_T;           // 4096 token rows

  // Workspace carve-out (256B aligned). Total ~284MB.
  char* wsb = (char*)d_ws;
  size_t off = 0;
  auto take = [&](size_t bytes) -> void* {
    void* p = wsb + off;
    off += (bytes + 255) & ~(size_t)255;
    return p;
  };
  bf16* wqkv = (bf16*)take((size_t)WG_L * 3 * WG_D * WG_D * 2);
  bf16* wprj = (bf16*)take((size_t)WG_L * WG_D * WG_D * 2);
  bf16* wfc1 = (bf16*)take((size_t)WG_L * WG_FF * WG_D * 2);
  bf16* wfc2 = (bf16*)take((size_t)WG_L * WG_D * WG_FF * 2);
  bf16* wemb = (bf16*)take((size_t)WG_VOCAB * WG_D * 2);
  float* hf  = (float*)take((size_t)M * WG_D * 4);          // fp32 residual
  bf16* xb   = (bf16*) take((size_t)M * WG_D * 2);          // LN output
  bf16* qkvb = (bf16*) take((size_t)M * 3 * WG_D * 2);
  bf16* qh   = (bf16*) take((size_t)M * WG_D * 2);          // [B,H,T,HD]
  bf16* kh   = (bf16*) take((size_t)M * WG_D * 2);
  bf16* vh   = (bf16*) take((size_t)M * WG_D * 2);
  bf16* attn = (bf16*) take((size_t)M * WG_D * 2);
  bf16* mb   = (bf16*) take((size_t)M * WG_FF * 2);

  // 1) Weights -> bf16 (deterministic each launch; hot in L2 afterwards)
  f2bf_kernel<<<2048, 256, 0, stream>>>(qkv_w, wqkv, (size_t)WG_L * 3 * WG_D * WG_D);
  f2bf_kernel<<<2048, 256, 0, stream>>>(proj_w, wprj, (size_t)WG_L * WG_D * WG_D);
  f2bf_kernel<<<2048, 256, 0, stream>>>(fc1_w, wfc1, (size_t)WG_L * WG_FF * WG_D);
  f2bf_kernel<<<2048, 256, 0, stream>>>(fc2_w, wfc2, (size_t)WG_L * WG_D * WG_FF);
  f2bf_kernel<<<2048, 256, 0, stream>>>(tok_emb, wemb, (size_t)WG_VOCAB * WG_D);

  // 2) Embedding
  embed_kernel<<<M, 256, 0, stream>>>(x, tok_emb, pos_emb, hf);

  auto gemm = [&](const bf16* A, const bf16* W, const float* bias, float* resid,
                  bf16* ob, float* of, int Mm, int Nn, int Kk, int gelu) {
    dim3 grid(Nn / 128, Mm / 128);
    gemm_kernel<<<grid, 256, 0, stream>>>(A, W, bias, resid, ob, of, Mm, Nn, Kk, gelu);
  };

  // 3) Transformer layers
  for (int l = 0; l < WG_L; ++l) {
    ln_kernel<<<M, 256, 0, stream>>>(hf, ln1_g + l * WG_D, ln1_b + l * WG_D, xb);
    gemm(xb, wqkv + (size_t)l * 3 * WG_D * WG_D, qkv_b + l * 3 * WG_D,
         nullptr, qkvb, nullptr, M, 3 * WG_D, WG_D, 0);
    qkv_split_kernel<<<(M * 3 * WG_D) / 256, 256, 0, stream>>>(qkvb, qh, kh, vh);
    attn_kernel<<<dim3(WG_T / 128, WG_B * WG_H), 256, 0, stream>>>(qh, kh, vh, attn);
    gemm(attn, wprj + (size_t)l * WG_D * WG_D, proj_b + l * WG_D,
         hf, nullptr, nullptr, M, WG_D, WG_D, 0);                 // += residual
    ln_kernel<<<M, 256, 0, stream>>>(hf, ln2_g + l * WG_D, ln2_b + l * WG_D, xb);
    gemm(xb, wfc1 + (size_t)l * WG_FF * WG_D, fc1_b + l * WG_FF,
         nullptr, mb, nullptr, M, WG_FF, WG_D, 1);                // GELU
    gemm(mb, wfc2 + (size_t)l * WG_D * WG_FF, fc2_b + l * WG_D,
         hf, nullptr, nullptr, M, WG_D, WG_FF, 0);                // += residual
  }

  // 4) Final LN + logits (fp32 out, N=32000 = 250*128)
  ln_kernel<<<M, 256, 0, stream>>>(hf, lnf_g, lnf_b, xb);
  gemm(xb, wemb, nullptr, nullptr, nullptr, logits, M, WG_VOCAB, WG_D, 0);
}
